// VQBaseline_23124103922125
// MI455X (gfx1250) — matrix operand order repaired and checked by reading, not compiled, not run on hardware
//
#include <hip/hip_runtime.h>

typedef __attribute__((ext_vector_type(16))) __bf16 v16bf;
typedef __attribute__((ext_vector_type(8)))  float  v8f;

#define DIM  256
#define KCB  4096   // codebook entries

// ---------------- ws layout ----------------
// bhi   : KCB*DIM bf16  (2 MB)   pre-swizzled B-matrix hi plane
// blo   : KCB*DIM bf16  (2 MB)   pre-swizzled B-matrix lo plane
// cnorm : KCB float     (16 KB)  exact ||c||^2

// Pre-swizzle codebook into WMMA B-layout (32x16 bf16 per tile/k-step):
// block g = (code_tile ct)*8 + kstep ks ; within block:
//   lane L: code = ct*16 + L%16, dims = ks*32 + (L<16?0:16) + 0..15
__global__ __launch_bounds__(256) void vq_prep_b(const float* __restrict__ cb,
                                                 __bf16* __restrict__ bhi,
                                                 __bf16* __restrict__ blo) {
  const int tid  = blockIdx.x * 256 + threadIdx.x;   // 65536 threads total
  const int lane = tid & 31;
  const int g    = tid >> 5;         // 0..2047
  const int ct   = g >> 3;           // code tile 0..255
  const int ks   = g & 7;            // k-step 0..7
  const int half = lane >> 4;
  const int l16  = lane & 15;
  const int code = ct * 16 + l16;
  const int doff = ks * 32 + half * 16;

  const float* bp = cb + (size_t)code * DIM + doff;
  float4 b0 = *(const float4*)(bp);
  float4 b1 = *(const float4*)(bp + 4);
  float4 b2 = *(const float4*)(bp + 8);
  float4 b3 = *(const float4*)(bp + 12);
  float bv[16] = {b0.x,b0.y,b0.z,b0.w, b1.x,b1.y,b1.z,b1.w,
                  b2.x,b2.y,b2.z,b2.w, b3.x,b3.y,b3.z,b3.w};
  v16bf bh, bl;
#pragma unroll
  for (int i = 0; i < 16; ++i) {
    __bf16 h = (__bf16)bv[i];
    bh[i] = h;
    bl[i] = (__bf16)(bv[i] - (float)h);
  }
  const size_t eoff = ((size_t)g * 32 + lane) * 16;
  *(v16bf*)(bhi + eoff) = bh;
  *(v16bf*)(blo + eoff) = bl;
}

// Exact fp32 ||c||^2, one wave per code.
__global__ __launch_bounds__(256) void vq_cnorm(const float* __restrict__ cb,
                                                float* __restrict__ cnorm) {
  const int lane = threadIdx.x & 31;
  const int wave = threadIdx.x >> 5;
  const int code = blockIdx.x * 8 + wave;
  const float* p = cb + (size_t)code * DIM + lane * 8;
  float4 a = *(const float4*)p;
  float4 b = *(const float4*)(p + 4);
  float s = a.x*a.x + a.y*a.y + a.z*a.z + a.w*a.w
          + b.x*b.x + b.y*b.y + b.z*b.z + b.w*b.w;
#pragma unroll
  for (int off = 16; off; off >>= 1) s += __shfl_xor(s, off, 32);
  if (lane == 0) cnorm[code] = s;
}

// Main: 8 waves/WG, 16 rows/wave (128 rows/WG). A operands converted once per
// wave into LDS (pre-swizzled hi/lo planes, 16KB/wave); B operands loaded
// pre-converted from ws; 3 bf16 WMMAs per 16x16 tile per 32-dim k-step.
__global__ __launch_bounds__(256) void vq_main(
    const float* __restrict__ x,
    const float* __restrict__ cb,
    const __bf16* __restrict__ bhi,
    const __bf16* __restrict__ blo,
    const float* __restrict__ cnorm,
    float* __restrict__ out) {

  extern __shared__ char smem[];     // 128 KB: 16 KB per wave

  const int lane = threadIdx.x & 31;
  const int wave = threadIdx.x >> 5;
  const int half = lane >> 4;
  const int l16  = lane & 15;

  const int rowBase = blockIdx.x * 128 + wave * 16;
  const int aRow    = rowBase + l16;   // A layout: M = lane%16 in both halves

  char* aHi = smem + wave * 16384 + lane * 32;   // [ks]: +ks*1024
  char* aLo = aHi + 8192;

  // ---- convert this wave's A tile once into LDS (hi/lo planes) ----
#pragma unroll
  for (int ks = 0; ks < 8; ++ks) {
    // lanes 0-15: K = kd+0..15 ; lanes 16-31: K = kd+16..31
    const float* ap = x + (size_t)aRow * DIM + ks * 32 + half * 16;
    float4 a0 = *(const float4*)(ap);
    float4 a1 = *(const float4*)(ap + 4);
    float4 a2 = *(const float4*)(ap + 8);
    float4 a3 = *(const float4*)(ap + 12);
    float av[16] = {a0.x,a0.y,a0.z,a0.w, a1.x,a1.y,a1.z,a1.w,
                    a2.x,a2.y,a2.z,a2.w, a3.x,a3.y,a3.z,a3.w};
    v16bf ah, al;
#pragma unroll
    for (int i = 0; i < 16; ++i) {
      __bf16 h = (__bf16)av[i];
      ah[i] = h;
      al[i] = (__bf16)(av[i] - (float)h);
    }
    *(v16bf*)(aHi + ks * 1024) = ah;
    *(v16bf*)(aLo + ks * 1024) = al;
  }

  float best[8];
  int   bidx[8];
#pragma unroll
  for (int j = 0; j < 8; ++j) { best[j] = 3.402823466e38f; bidx[j] = 0; }

  for (int kc = 0; kc < KCB; kc += 64) {
    const int ct0 = kc >> 4;         // first of 4 code tiles this chunk
    v8f acc0 = {}, acc1 = {}, acc2 = {}, acc3 = {};

#pragma unroll 4
    for (int ks = 0; ks < 8; ++ks) {
      const v16bf ah = *(const v16bf*)(aHi + ks * 1024);
      const v16bf al = *(const v16bf*)(aLo + ks * 1024);
#define DO_TILE(T, ACC)                                                        \
      {                                                                        \
        const size_t eoff = (((size_t)(ct0 + (T)) * 8 + ks) * 32 + lane) * 16; \
        v16bf bh = *(const v16bf*)(bhi + eoff);                                \
        v16bf bl = *(const v16bf*)(blo + eoff);                                \
        ACC = __builtin_amdgcn_wmma_f32_16x16x32_bf16(false, ah, false, bh,    \
                                                      (short)0, ACC, false, false); \
        ACC = __builtin_amdgcn_wmma_f32_16x16x32_bf16(false, al, false, bh,    \
                                                      (short)0, ACC, false, false); \
        ACC = __builtin_amdgcn_wmma_f32_16x16x32_bf16(false, ah, false, bl,    \
                                                      (short)0, ACC, false, false); \
      }
      DO_TILE(0, acc0)
      DO_TILE(1, acc1)
      DO_TILE(2, acc2)
      DO_TILE(3, acc3)
#undef DO_TILE
    }

    // fold chunk into running argmin: dist = ||c||^2 - 2*x.c
#define FIN_TILE(T, ACC)                                                       \
    {                                                                          \
      const int   code = kc + (T) * 16 + l16;                                  \
      const float cnF  = cnorm[code];                                          \
      _Pragma("unroll")                                                        \
      for (int j = 0; j < 8; ++j) {                                            \
        float d = cnF - 2.0f * (ACC)[j];                                       \
        if (d < best[j]) { best[j] = d; bidx[j] = code; }                      \
      }                                                                        \
    }
    FIN_TILE(0, acc0)
    FIN_TILE(1, acc1)
    FIN_TILE(2, acc2)
    FIN_TILE(3, acc3)
#undef FIN_TILE
  }

  // ---- reduce across the 16 lanes sharing each row (per half-wave) ----
#pragma unroll
  for (int j = 0; j < 8; ++j) {
#pragma unroll
    for (int off = 8; off; off >>= 1) {
      float od = __shfl_xor(best[j], off, 32);
      int   oi = __shfl_xor(bidx[j], off, 32);
      if (od < best[j] || (od == best[j] && oi < bidx[j])) {
        best[j] = od; bidx[j] = oi;
      }
    }
  }

  // ---- gather: out[row] = codebook[argmin[row]] (exact fp32 rows) ----
#pragma unroll
  for (int j = 0; j < 8; ++j) {
    int idxA = __shfl(bidx[j], 0, 32);   // rows 0-7  (lanes 0-15)
    int idxB = __shfl(bidx[j], 16, 32);  // rows 8-15 (lanes 16-31)
    {
      const float4* src = (const float4*)(cb + (size_t)idxA * DIM);
      float4* dst = (float4*)(out + (size_t)(rowBase + j) * DIM);
      dst[lane]      = src[lane];
      dst[lane + 32] = src[lane + 32];
    }
    {
      const float4* src = (const float4*)(cb + (size_t)idxB * DIM);
      float4* dst = (float4*)(out + (size_t)(rowBase + 8 + j) * DIM);
      dst[lane]      = src[lane];
      dst[lane + 32] = src[lane + 32];
    }
  }
}

extern "C" void kernel_launch(void* const* d_in, const int* in_sizes, int n_in,
                              void* d_out, int out_size, void* d_ws, size_t ws_size,
                              hipStream_t stream) {
  const float* x  = (const float*)d_in[0];   // [N, D] fp32
  const float* cb = (const float*)d_in[1];   // [K, D] fp32
  float* out = (float*)d_out;                // [N, D] fp32

  __bf16* bhi  = (__bf16*)d_ws;
  __bf16* blo  = bhi + (size_t)KCB * DIM;
  float*  cnrm = (float*)(blo + (size_t)KCB * DIM);

  const int n    = in_sizes[0] / DIM;        // 32768
  const int grid = n / 128;                  // 128 rows per workgroup

  vq_prep_b<<<256, 256, 0, stream>>>(cb, bhi, blo);          // 65536 threads
  vq_cnorm <<<KCB / 8, 256, 0, stream>>>(cb, cnrm);          // wave per code
  vq_main  <<<grid, 256, 131072, stream>>>(x, cb, bhi, blo, cnrm, out);
}
